// Generator_58274116272542
// MI455X (gfx1250) — matrix operand order repaired and checked by reading, not compiled, not run on hardware
//
#include <hip/hip_runtime.h>

typedef __bf16 bf16;
typedef __attribute__((ext_vector_type(8)))  bf16  v8bf;
typedef __attribute__((ext_vector_type(16))) bf16  v16bf;
typedef __attribute__((ext_vector_type(4)))  float v4f;
typedef __attribute__((ext_vector_type(8)))  float v8f;

constexpr int N    = 32768;     // batch
constexpr int DIN  = 1024;      // embedding dim
constexpr int HID  = 1024;      // hidden dim
constexpr int INTD = 2048;      // internal dim
constexpr int NPAD = N + 128;   // permuted row space (gap-padded, 128-aligned split)
constexpr int MB   = 64;        // rows per block (4 WMMA m-tiles)
constexpr int MT   = MB / 16;   // m-tiles per block
constexpr int NQ   = 4;         // n-tiles per wave iteration (quad)
constexpr int NTB  = NPAD / MB; // 514 row blocks

// Swizzled (fragment-major) LDS A-tile layout, element units (bf16):
//   idx(R, k) = (k>>5)*2048 + (R>>4)*512 + ((k>>4)&1)*256
//             + (((k>>3)&1)*16 + (R&15))*8 + (k&7)
// All DS addressing = one lane base + immediates; lanes bank-dense.

__device__ __forceinline__ bf16 f2bf(float f) {
  unsigned u = __builtin_bit_cast(unsigned, f);
  unsigned r = u + 0x7FFFu + ((u >> 16) & 1u);   // round-to-nearest-even
  unsigned short s = (unsigned short)(r >> 16);
  return __builtin_bit_cast(bf16, s);
}

union Frag { v16bf v; v8bf h[2]; };

__device__ __forceinline__ v8f wmma_bf16(const Frag& a, const Frag& b, v8f c) {
  return __builtin_amdgcn_wmma_f32_16x16x32_bf16(
      false, a.v, false, b.v, (short)0, c, false, false);
}

// ---------------- routing kernels ----------------
__global__ void init_kernel(int* meta, int* perm) {
  int i = blockIdx.x * blockDim.x + threadIdx.x;
  if (i < 64) meta[i] = 0;
  if (i < NPAD) perm[i] = -1;
}
__global__ void count_kernel(const int* __restrict__ lang, int* meta) {
  int i = blockIdx.x * blockDim.x + threadIdx.x;
  if (i < N) atomicAdd(&meta[lang[i] != 0 ? 1 : 0], 1);
}
__global__ void base_kernel(int* meta) {
  if (blockIdx.x == 0 && threadIdx.x == 0) meta[4] = (meta[0] + 127) & ~127;
}
__global__ void fill_kernel(const int* __restrict__ lang, int* meta, int* perm) {
  int i = blockIdx.x * blockDim.x + threadIdx.x;
  if (i < N) {
    int l = (lang[i] != 0) ? 1 : 0;
    int pos = atomicAdd(&meta[2 + l], 1);
    perm[l ? (meta[4] + pos) : pos] = i;
  }
}

// ------- fp32 -> bf16 transposed weight conversion (coalesced both sides) ---
// in: W[k][n] (K x Nout), out: Wt[n][k] (Nout x K). grid = (Nout/32, K/32)
__global__ __launch_bounds__(256) void cvt_kernel(const float* __restrict__ W,
                                                  bf16* __restrict__ Wt,
                                                  int K, int Nout) {
  __shared__ float t[32][33];
  const int nBase = blockIdx.x * 32, kBase = blockIdx.y * 32;
  const int tx = threadIdx.x & 31, ty = threadIdx.x >> 5;
  #pragma unroll
  for (int j = 0; j < 32; j += 8)
    t[ty + j][tx] = W[(long)(kBase + ty + j) * Nout + nBase + tx];
  __syncthreads();
  #pragma unroll
  for (int j = 0; j < 32; j += 8)
    Wt[(long)(nBase + ty + j) * K + kBase + tx] = f2bf(t[tx][ty + j]);
}

// ------- shared preamble: load perm slice, early-out on all-gap tiles -------
__device__ __forceinline__ bool tile_prologue(const int* __restrict__ perm,
                                              int rowBase, int* sPerm, int* sAny) {
  const int tid = threadIdx.x;
  if (tid == 0) *sAny = 0;
  __syncthreads();
  if (tid < MB) {
    int p = perm[rowBase + tid];
    sPerm[tid] = p;
    if (p >= 0) *sAny = 1;   // benign race
  }
  __syncthreads();
  return *sAny != 0;
}

// ---- quad-n-tile GEMM core over a swizzled A-tile --------------------
// One A-fragment feeds 4 WMMAs (n-quad -> 0.5 ds_load/WMMA, LDS balanced
// with XDL), one B-fragment feeds 4 WMMAs (m-tile reuse).
template <int K, typename Epi>
__device__ __forceinline__ void gemm4(const bf16* __restrict__ sA,
                                      const bf16* __restrict__ Wt,
                                      int nTiles, int wave, int nl, int half,
                                      Epi epi) {
  const bf16* ab = sA + (threadIdx.x & 31) * 8;  // single DS base register

  for (int nq = wave; nq < nTiles / NQ; nq += 8) {
    const int nG0 = nq * (NQ * 16) + nl;
    const bf16* w[NQ];
    #pragma unroll
    for (int q = 0; q < NQ; ++q)
      w[q] = Wt + (long)(nG0 + q * 16) * K + half * 8;
    __builtin_prefetch(Wt + (long)(nG0 + 8 * NQ * 16) * K, 0, 0);

    v8f acc[MT][NQ] = {};
    #pragma unroll 2
    for (int kb = 0; kb < K / 32; ++kb) {
      Frag b[NQ];
      #pragma unroll
      for (int q = 0; q < NQ; ++q) {
        b[q].h[0] = *(const v8bf*)(w[q] + kb * 32);
        b[q].h[1] = *(const v8bf*)(w[q] + kb * 32 + 16);
      }
      #pragma unroll
      for (int mt = 0; mt < MT; ++mt) {
        Frag a;
        const bf16* p = ab + kb * 2048 + mt * 512;  // immediates only
        a.h[0] = *(const v8bf*)(p);
        a.h[1] = *(const v8bf*)(p + 256);
        #pragma unroll
        for (int q = 0; q < NQ; ++q)
          acc[mt][q] = wmma_bf16(a, b[q], acc[mt][q]);
      }
    }
    epi(nq, nG0, acc);
  }
}

// stage a row-major bf16 global tile into the swizzled LDS layout
template <int K>
__device__ __forceinline__ void stage_swz(bf16* sA, const bf16* __restrict__ src,
                                          int rowBase) {
  constexpr int CPR = K / 8;  // 16B chunks per row
  for (int i = threadIdx.x; i < MB * CPR; i += 256) {
    int R = i / CPR, j = i % CPR;
    int kb = j >> 2, chunk = (j >> 1) & 1, hf = j & 1;
    int dst = kb * 2048 + (R >> 4) * 512 + chunk * 256 + (hf * 16 + (R & 15)) * 8;
    *(v8bf*)(sA + dst) =
        *(const v8bf*)(src + (long)(rowBase + R) * K + kb * 32 + chunk * 16 + hf * 8);
  }
}

// ---------------- fused expert encoder: relu(x@W1+b1)@W2+b2 ----------------
__global__ __launch_bounds__(256) void encoder_kernel(
    const float* __restrict__ x, const int* __restrict__ perm,
    const int* __restrict__ meta,
    const bf16* __restrict__ W1a, const float* __restrict__ b1a,
    const bf16* __restrict__ W2a, const float* __restrict__ b2a,
    const bf16* __restrict__ W1b, const float* __restrict__ b1b,
    const bf16* __restrict__ W2b, const float* __restrict__ b2b,
    bf16* __restrict__ enc) {
  extern __shared__ char smem[];
  bf16* sX    = (bf16*)smem;             // [MB*DIN] swizzled
  bf16* sH    = sX + MB * DIN;           // [MB*HID] swizzled
  int*  sPerm = (int*)(sH + MB * HID);   // [MB]
  int*  sAny  = sPerm + MB;

  const int tid = threadIdx.x;
  const int rowBase = blockIdx.x * MB;
  if (!tile_prologue(perm, rowBase, sPerm, sAny)) return;

  const bool e1 = (rowBase >= meta[4]);  // 128-aligned split: blocks expert-pure
  const bf16*  W1 = e1 ? W1b : W1a;  const float* B1 = e1 ? b1b : b1a;
  const bf16*  W2 = e1 ? W2b : W2a;  const float* B2 = e1 ? b2b : b2a;

  // gather x rows -> swizzled LDS bf16 (8 floats -> one 16B fragment chunk)
  for (int i = tid; i < MB * (DIN / 8); i += 256) {
    int R = i >> 7, j = i & 127;
    int kb = j >> 2, chunk = (j >> 1) & 1, hf = j & 1;
    int g = sPerm[R];
    int kBase = kb * 32 + chunk * 16 + hf * 8;
    v4f lo = {}, hi = {};
    if (g >= 0) {
      lo = *(const v4f*)(x + (long)g * DIN + kBase);
      hi = *(const v4f*)(x + (long)g * DIN + kBase + 4);
    }
    v8bf o;
    #pragma unroll
    for (int q = 0; q < 4; ++q) { o[q] = f2bf(lo[q]); o[q + 4] = f2bf(hi[q]); }
    int dst = kb * 2048 + (R >> 4) * 512 + chunk * 256 + (hf * 16 + (R & 15)) * 8;
    *(v8bf*)(sX + dst) = o;
  }
  __syncthreads();

  const int wave = tid >> 5, lane = tid & 31;
  const int nl = lane & 15, half = lane >> 4, m0 = half * 8;
  const int hfc = nl >> 3, el = nl & 7;   // column position inside fragment chunk

  // layer 1: h = relu(x @ W1 + b1), written directly in swizzled form
  gemm4<DIN>(sX, W1, HID / 16, wave, nl, half,
      [&](int nq, int nG0, v8f (&acc)[MT][NQ]) {
        #pragma unroll
        for (int q = 0; q < NQ; ++q) {
          float bb = B1[nG0 + q * 16];
          int kbq = nq * 2 + (q >> 1);
          int chk = (q & 1) * 256;
          #pragma unroll
          for (int mt = 0; mt < MT; ++mt)
            #pragma unroll
            for (int i = 0; i < 8; ++i) {
              float v = acc[mt][q][i] + bb;
              int idx = kbq * 2048 + mt * 512 + chk + (hfc * 16 + m0 + i) * 8 + el;
              sH[idx] = f2bf(v > 0.0f ? v : 0.0f);
            }
        }
      });
  __syncthreads();

  // layer 2: enc = h @ W2 + b2  (bf16, row-major, permuted layout)
  gemm4<HID>(sH, W2, INTD / 16, wave, nl, half,
      [&](int nq, int nG0, v8f (&acc)[MT][NQ]) {
        (void)nq;
        #pragma unroll
        for (int q = 0; q < NQ; ++q) {
          float bb = B2[nG0 + q * 16];
          #pragma unroll
          for (int mt = 0; mt < MT; ++mt)
            #pragma unroll
            for (int i = 0; i < 8; ++i) {
              long row = rowBase + mt * 16 + m0 + i;
              enc[row * INTD + nG0 + q * 16] = f2bf(acc[mt][q][i] + bb);
            }
        }
      });
}

// -------- decoder stage 1: h = relu(enc @ Wd1 + b1)  (K=INTD, N=HID) --------
__global__ __launch_bounds__(256) void dec1_kernel(
    const bf16* __restrict__ enc, const int* __restrict__ perm,
    const bf16* __restrict__ W1, const float* __restrict__ B1,
    bf16* __restrict__ hbuf) {
  extern __shared__ char smem[];
  bf16* sE    = (bf16*)smem;             // [MB*INTD] swizzled
  int*  sPerm = (int*)(sE + MB * INTD);
  int*  sAny  = sPerm + MB;

  const int tid = threadIdx.x;
  const int rowBase = blockIdx.x * MB;
  if (!tile_prologue(perm, rowBase, sPerm, sAny)) return;

  stage_swz<INTD>(sE, enc, rowBase);
  __syncthreads();

  const int wave = tid >> 5, lane = tid & 31;
  const int nl = lane & 15, half = lane >> 4, m0 = half * 8;

  gemm4<INTD>(sE, W1, HID / 16, wave, nl, half,
      [&](int nq, int nG0, v8f (&acc)[MT][NQ]) {
        (void)nq;
        #pragma unroll
        for (int q = 0; q < NQ; ++q) {
          float bb = B1[nG0 + q * 16];
          #pragma unroll
          for (int mt = 0; mt < MT; ++mt)
            #pragma unroll
            for (int i = 0; i < 8; ++i) {
              float v = acc[mt][q][i] + bb;
              long row = rowBase + mt * 16 + m0 + i;
              hbuf[row * HID + nG0 + q * 16] = f2bf(v > 0.0f ? v : 0.0f);
            }
        }
      });
}

// ---- decoder stage 2: out = h @ Wd2 + b2, fp32 scatter via perm (K=HID) ----
__global__ __launch_bounds__(256) void dec2_kernel(
    const bf16* __restrict__ hbuf, const int* __restrict__ perm,
    const bf16* __restrict__ W2, const float* __restrict__ B2,
    float* __restrict__ out) {
  extern __shared__ char smem[];
  bf16* sH    = (bf16*)smem;             // [MB*HID] swizzled
  int*  sPerm = (int*)(sH + MB * HID);
  int*  sAny  = sPerm + MB;

  const int tid = threadIdx.x;
  const int rowBase = blockIdx.x * MB;
  if (!tile_prologue(perm, rowBase, sPerm, sAny)) return;

  stage_swz<HID>(sH, hbuf, rowBase);
  __syncthreads();

  const int wave = tid >> 5, lane = tid & 31;
  const int nl = lane & 15, half = lane >> 4, m0 = half * 8;

  gemm4<HID>(sH, W2, DIN / 16, wave, nl, half,
      [&](int nq, int nG0, v8f (&acc)[MT][NQ]) {
        (void)nq;
        #pragma unroll
        for (int mt = 0; mt < MT; ++mt)
          #pragma unroll
          for (int i = 0; i < 8; ++i) {
            int g = sPerm[mt * 16 + m0 + i];
            if (g >= 0) {
              #pragma unroll
              for (int q = 0; q < NQ; ++q)
                out[(long)g * DIN + nG0 + q * 16] = acc[mt][q][i] + B2[nG0 + q * 16];
            }
          }
      });
}

extern "C" void kernel_launch(void* const* d_in, const int* in_sizes, int n_in,
                              void* d_out, int out_size, void* d_ws, size_t ws_size,
                              hipStream_t stream) {
  (void)in_sizes; (void)n_in; (void)out_size; (void)ws_size;
  const float* x    = (const float*)d_in[0];
  const int*   lang = (const int*)d_in[1];
  const float* Wde1 = (const float*)d_in[2];
  const float* bde1 = (const float*)d_in[3];
  const float* Wde2 = (const float*)d_in[4];
  const float* bde2 = (const float*)d_in[5];
  const float* Wnl1 = (const float*)d_in[6];
  const float* bnl1 = (const float*)d_in[7];
  const float* Wnl2 = (const float*)d_in[8];
  const float* bnl2 = (const float*)d_in[9];
  const float* Wd1  = (const float*)d_in[10];
  const float* bd1  = (const float*)d_in[11];
  const float* Wd2  = (const float*)d_in[12];
  const float* bd2  = (const float*)d_in[13];
  float* out = (float*)d_out;

  // workspace layout
  char* ws = (char*)d_ws;
  size_t off = 0;
  int* meta = (int*)(ws + off); off += 256;
  int* perm = (int*)(ws + off); off += (size_t)NPAD * 4;           // 256-aligned
  bf16* Wde1t = (bf16*)(ws + off); off += (size_t)HID  * DIN  * 2; // [HID][DIN]
  bf16* Wde2t = (bf16*)(ws + off); off += (size_t)INTD * HID  * 2; // [INTD][HID]
  bf16* Wnl1t = (bf16*)(ws + off); off += (size_t)HID  * DIN  * 2;
  bf16* Wnl2t = (bf16*)(ws + off); off += (size_t)INTD * HID  * 2;
  bf16* Wd1t  = (bf16*)(ws + off); off += (size_t)HID  * INTD * 2; // [HID][INTD]
  bf16* Wd2t  = (bf16*)(ws + off); off += (size_t)DIN  * HID  * 2; // [DIN][HID]
  bf16* enc   = (bf16*)(ws + off); off += (size_t)NPAD * INTD * 2;
  bf16* hbuf  = (bf16*)(ws + off); off += (size_t)NPAD * HID  * 2;

  // 1) routing (expert-sorted permutation, 128-aligned group split)
  init_kernel <<<(NPAD + 255) / 256, 256, 0, stream>>>(meta, perm);
  count_kernel<<<(N + 255) / 256, 256, 0, stream>>>(lang, meta);
  base_kernel <<<1, 32, 0, stream>>>(meta);
  fill_kernel <<<(N + 255) / 256, 256, 0, stream>>>(lang, meta, perm);

  // 2) weight conversion (fp32 -> bf16, transposed to [n][k], tiled/coalesced)
  cvt_kernel<<<dim3(HID / 32,  DIN / 32),  256, 0, stream>>>(Wde1, Wde1t, DIN,  HID);
  cvt_kernel<<<dim3(INTD / 32, HID / 32),  256, 0, stream>>>(Wde2, Wde2t, HID,  INTD);
  cvt_kernel<<<dim3(HID / 32,  DIN / 32),  256, 0, stream>>>(Wnl1, Wnl1t, DIN,  HID);
  cvt_kernel<<<dim3(INTD / 32, HID / 32),  256, 0, stream>>>(Wnl2, Wnl2t, HID,  INTD);
  cvt_kernel<<<dim3(HID / 32,  INTD / 32), 256, 0, stream>>>(Wd1,  Wd1t,  INTD, HID);
  cvt_kernel<<<dim3(DIN / 32,  HID / 32),  256, 0, stream>>>(Wd2,  Wd2t,  HID,  DIN);

  // 3) fused routed encoder (64-row expert-pure blocks, 4x B / 4x A reuse)
  const unsigned encSmem = (MB * DIN + MB * HID) * sizeof(bf16) + (MB + 8) * sizeof(int);
  encoder_kernel<<<NTB, 256, encSmem, stream>>>(
      x, perm, meta, Wde1t, bde1, Wde2t, bde2, Wnl1t, bnl1, Wnl2t, bnl2, enc);

  // 4) decoder stage 1 (K=2048), stage 2 (K=1024) + fp32 scatter
  const unsigned d1Smem = (MB * INTD) * sizeof(bf16) + (MB + 8) * sizeof(int);
  dec1_kernel<<<NTB, 256, d1Smem, stream>>>(enc, perm, Wd1t, bd1, hbuf);
  const unsigned d2Smem = (MB * HID) * sizeof(bf16) + (MB + 8) * sizeof(int);
  dec2_kernel<<<NTB, 256, d2Smem, stream>>>(hbuf, perm, Wd2t, bd2, out);
}
